// NTreeMGU_80719615361096
// MI455X (gfx1250) — matrix operand order repaired and checked by reading, not compiled, not run on hardware
//
#include <hip/hip_runtime.h>
#include <math.h>

// ---------------------------------------------------------------------------
// NTreeMGU on MI455X (gfx1250): WMMA bf16 tile-GEMM core + fused epilogues,
// with GLOBAL_LOAD_ASYNC_TO_LDS_B128 staging (ASYNCcnt pipeline).
//
//   wx   = x @ W_w^T + W_b                  (131071x256)·(256x768)
//   leaves: h = (1 - Σ sigmoid(wfx)) * tanh(whcx)
//   per level d = 15..0 (cnt = 2^d):
//     f_pre  = h_cat @ U_f^T                (cnt x 512)·(512x512)
//     fh     = f_pre * h_cat
//     hc_add = fh @ U_hc^T                  (cnt x 512)·(512x256)
//     h_new  = fh[:, :H]+fh[:, H:] + (1-Σ sigmoid(f_pre+wfx)) * tanh(whcx+hc_add)
//
// GEMM operands are pre-converted to bf16 once (x, W_w, U_f, U_hc) or emitted
// as bf16 mirrors by the epilogues (h, fh); staging is then a pure bf16 copy
// done with async global->LDS b128 ops, double-buffered, overlapped with
// v_wmma_f32_16x16x32_bf16 (fp32 accumulate). Elementwise math stays fp32.
// ---------------------------------------------------------------------------

typedef __bf16 bf16;
typedef __attribute__((address_space(3))) __bf16 lds_bf16;
typedef __attribute__((ext_vector_type(16))) __bf16 v16bf;
typedef __attribute__((ext_vector_type(8)))  __bf16 v8bf;
typedef __attribute__((ext_vector_type(8)))  float  v8f;

#define Hdim     256
#define Xdim     256
#define TDEPTH   16
#define NNODES   131071           // 2^17 - 1
#define WXC      (3 * Hdim)       // 768
#define LEAF0    65535            // 2^16 - 1

// tile configuration
#define TM      64
#define TN      128
#define TK      32
#define BLK     256               // 8 waves of 32
#define APITCH  40                // 80 B row pitch (16B multiple, bank-spread)
#define BPITCH  40

__device__ __forceinline__ float sigf(float v) {
    return 1.0f / (1.0f + __expf(-v));
}

__device__ __forceinline__ v8bf cvt8(const float* __restrict__ s) {
    const float4* s4 = (const float4*)s;
    float4 p0 = s4[0], p1 = s4[1];
    v8bf v;
    v[0] = (bf16)p0.x; v[1] = (bf16)p0.y; v[2] = (bf16)p0.z; v[3] = (bf16)p0.w;
    v[4] = (bf16)p1.x; v[5] = (bf16)p1.y; v[6] = (bf16)p1.z; v[7] = (bf16)p1.w;
    return v;
}

// 16-byte async copy: global -> LDS (ASYNCcnt-tracked, no VGPR data path)
__device__ __forceinline__ void async_cp_b128(void* lds_dst, const void* gsrc) {
    lds_bf16* lp = (lds_bf16*)lds_dst;           // 32-bit LDS address
    asm volatile("global_load_async_to_lds_b128 %0, %1, off"
                 :: "v"(lp), "v"(gsrc) : "memory");
}
__device__ __forceinline__ void wait_async0() {
    asm volatile("s_wait_asynccnt 0" ::: "memory");
}

// Stage one K-tile (bf16 -> bf16 pure copy, 3 async b128 per thread).
//   As[row][k] : 64 x 32 bf16 row-major          (A fragment = 1 contig run)
//   BsT[n][k]  : 128 x 32 bf16 K-contig per n    (B fragment = 1 contig run)
__device__ __forceinline__ void stage_async(bf16 (*__restrict__ As)[APITCH],
                                            bf16 (*__restrict__ BsT)[BPITCH],
                                            const bf16* __restrict__ A, int ldA,
                                            int Mtot, int m0,
                                            const bf16* __restrict__ B, int ldB,
                                            int n0, int k0) {
    const int t = threadIdx.x;
    {   // A: thread -> (row = t>>2, kc = (t&3)*8), one 16B chunk
        int row  = t >> 2;
        int kc   = (t & 3) * 8;
        int grow = m0 + row;
        if (grow >= Mtot) grow = Mtot - 1;       // clamp: rows guarded at store
        async_cp_b128(&As[row][kc], A + (size_t)grow * ldA + k0 + kc);
    }
    {   // B: thread -> (n = t>>1, kc = (t&1)*16), two 16B chunks
        int n  = t >> 1;
        int kc = (t & 1) * 16;
        const bf16* gp = B + (size_t)(n0 + n) * ldB + k0 + kc;
        async_cp_b128(&BsT[n][kc],     gp);
        async_cp_b128(&BsT[n][kc + 8], gp + 8);
    }
}

// ---------------------------------------------------------------------------
// Shared WMMA mainloop: acc += A[m0:m0+64, :] @ B[n0:n0+128, :]^T  (bf16)
// ---------------------------------------------------------------------------
__device__ __forceinline__ void gemm_bf16_tile(const bf16* __restrict__ A, int ldA, int Mtot,
                                               const bf16* __restrict__ B, int ldB,
                                               int m0, int n0, int Kdim, v8f acc[4]) {
    __shared__ alignas(16) bf16 As[2][TM][APITCH];
    __shared__ alignas(16) bf16 Bs[2][TN][BPITCH];

    const int t    = threadIdx.x;
    const int lane = t & 31;
    const int wave = t >> 5;
    const int r    = lane & 15;         // fragment row / output col
    const int hb   = lane >> 4;         // half-wave selector
    const int wm   = wave & 3;          // wave M subtile (16 rows)
    const int wn   = wave >> 2;         // wave N subtile (64 cols)

    const int nK = Kdim / TK;

    stage_async(As[0], Bs[0], A, ldA, Mtot, m0, B, ldB, n0, 0);
    wait_async0();
    __syncthreads();

    for (int kk = 0; kk < nK; ++kk) {
        const int buf = kk & 1;
        if (kk + 1 < nK)   // async-prefetch next K-tile into the other buffer
            stage_async(As[buf ^ 1], Bs[buf ^ 1], A, ldA, Mtot, m0,
                        B, ldB, n0, (kk + 1) * TK);

        // A fragment: K run {16hb..16hb+15} (2x b128)
        const bf16* ap = &As[buf][wm * 16 + r][0];
        v8bf a0 = *(const v8bf*)(ap + 16 * hb);
        v8bf a1 = *(const v8bf*)(ap + 16 * hb + 8);
        v16bf af = __builtin_shufflevector(a0, a1, 0, 1, 2, 3, 4, 5, 6, 7,
                                           8, 9, 10, 11, 12, 13, 14, 15);
        // Load all four B fragments first, then issue the four WMMAs.
        v16bf bfr[4];
#pragma unroll
        for (int tt = 0; tt < 4; ++tt) {
            const bf16* bp = &Bs[buf][wn * 64 + tt * 16 + r][16 * hb];
            v8bf b0 = *(const v8bf*)bp;
            v8bf b1 = *(const v8bf*)(bp + 8);
            bfr[tt] = __builtin_shufflevector(b0, b1, 0, 1, 2, 3, 4, 5, 6, 7,
                                              8, 9, 10, 11, 12, 13, 14, 15);
        }
#pragma unroll
        for (int tt = 0; tt < 4; ++tt)
            acc[tt] = __builtin_amdgcn_wmma_f32_16x16x32_bf16(
                false, af, false, bfr[tt], (short)0, acc[tt], false, false);

        wait_async0();       // our next-tile copies have landed in LDS
        __syncthreads();
    }
}

// Epilogue index helpers (C/D fragment layout: M = g + 8*hb, N = r)
#define EPILOGUE_SETUP()                         \
    const int t    = threadIdx.x;                \
    const int lane = t & 31;                     \
    const int wave = t >> 5;                     \
    const int r    = lane & 15;                  \
    const int hb   = lane >> 4;                  \
    const int wm   = wave & 3;                   \
    const int wn   = wave >> 2;                  \
    (void)t;

// ---------------------------------------------------------------------------
// Kernel 0: fp32 -> bf16 bulk conversion (8 elements / thread)
// ---------------------------------------------------------------------------
__global__ void __launch_bounds__(BLK)
k_cvt(const float* __restrict__ src, bf16* __restrict__ dst, int n8) {
    int i = blockIdx.x * BLK + threadIdx.x;
    if (i < n8)
        *(v8bf*)(dst + (size_t)i * 8) = cvt8(src + (size_t)i * 8);
}

// ---------------------------------------------------------------------------
// Kernel 1: wx = x @ W_w^T + W_b
// ---------------------------------------------------------------------------
__global__ void __launch_bounds__(BLK)
k_wx(const bf16* __restrict__ xb, const bf16* __restrict__ Wwb,
     const float* __restrict__ Wb, float* __restrict__ wx) {
    const int m0 = blockIdx.x * TM;
    const int n0 = blockIdx.y * TN;
    v8f acc[4] = {};
    gemm_bf16_tile(xb, Xdim, NNODES, Wwb, Xdim, m0, n0, Xdim, acc);

    EPILOGUE_SETUP();
#pragma unroll
    for (int tt = 0; tt < 4; ++tt) {
#pragma unroll
        for (int g = 0; g < 8; ++g) {
            int row = m0 + wm * 16 + g + 8 * hb;
            int col = n0 + wn * 64 + tt * 16 + r;
            if (row < NNODES)
                wx[(size_t)row * WXC + col] = acc[tt][g] + Wb[col];
        }
    }
}

// ---------------------------------------------------------------------------
// Kernel 2: leaf level — elementwise; emits fp32 h and bf16 mirror
// ---------------------------------------------------------------------------
__global__ void __launch_bounds__(Hdim)
k_leaf(const float* __restrict__ wx, float* __restrict__ h,
       bf16* __restrict__ hbm) {
    const int p = LEAF0 + blockIdx.x;
    const int i = threadIdx.x;
    const float* row = wx + (size_t)p * WXC;
    float whc = row[i];
    float fs  = sigf(row[Hdim + i]) + sigf(row[2 * Hdim + i]);
    float v   = (1.0f - fs) * tanhf(whc);
    h[(size_t)p * Hdim + i]   = v;
    hbm[(size_t)p * Hdim + i] = (bf16)v;
}

// ---------------------------------------------------------------------------
// Kernel 3: level GEMM #1 — f_pre = h_cat @ U_f^T ; fh = f_pre * h_cat
// h_cat is h[2*start+1 ...] viewed as (cnt x 512) row-major.
// ---------------------------------------------------------------------------
__global__ void __launch_bounds__(BLK)
k_fpre(const bf16* __restrict__ hbm, const float* __restrict__ h,
       const bf16* __restrict__ Ufb,
       float* __restrict__ f_pre, float* __restrict__ fh,
       bf16* __restrict__ fhb, int start, int cnt) {
    const bf16*  Ab = hbm + (size_t)(2 * start + 1) * Hdim;   // (cnt x 2H) bf16
    const float* Af = h   + (size_t)(2 * start + 1) * Hdim;   // (cnt x 2H) fp32
    const int m0 = blockIdx.x * TM;
    const int n0 = blockIdx.y * TN;
    v8f acc[4] = {};
    gemm_bf16_tile(Ab, 2 * Hdim, cnt, Ufb, 2 * Hdim, m0, n0, 2 * Hdim, acc);

    EPILOGUE_SETUP();
#pragma unroll
    for (int tt = 0; tt < 4; ++tt) {
#pragma unroll
        for (int g = 0; g < 8; ++g) {
            int row = m0 + wm * 16 + g + 8 * hb;
            int col = n0 + wn * 64 + tt * 16 + r;
            if (row < cnt) {
                size_t idx = (size_t)row * (2 * Hdim) + col;
                float  fp  = acc[tt][g];
                float  fhv = fp * Af[idx];
                f_pre[idx] = fp;
                fh[idx]    = fhv;
                fhb[idx]   = (bf16)fhv;
            }
        }
    }
}

// ---------------------------------------------------------------------------
// Kernel 4: level GEMM #2 — hc_add = fh @ U_hc^T, fused final combine:
//   h[p] = fh[:, :H]+fh[:, H:] + (1 - Σ sigmoid(f_pre + wfx)) * tanh(whcx + hc_add)
// ---------------------------------------------------------------------------
__global__ void __launch_bounds__(BLK)
k_hout(const bf16* __restrict__ fhb, const float* __restrict__ fh,
       const bf16* __restrict__ Uhcb,
       const float* __restrict__ f_pre, const float* __restrict__ wx,
       float* __restrict__ h, bf16* __restrict__ hbm, int start, int cnt) {
    const int m0 = blockIdx.x * TM;
    const int n0 = blockIdx.y * TN;
    v8f acc[4] = {};
    gemm_bf16_tile(fhb, 2 * Hdim, cnt, Uhcb, 2 * Hdim, m0, n0, 2 * Hdim, acc);

    EPILOGUE_SETUP();
#pragma unroll
    for (int tt = 0; tt < 4; ++tt) {
#pragma unroll
        for (int g = 0; g < 8; ++g) {
            int row = m0 + wm * 16 + g + 8 * hb;
            int i   = n0 + wn * 64 + tt * 16 + r;      // output col in [0, H)
            if (row < cnt) {
                int    p  = start + row;
                size_t rb = (size_t)row * (2 * Hdim);
                size_t wb = (size_t)p * WXC;
                float hcadd = acc[tt][g];
                float csum  = fh[rb + i] + fh[rb + Hdim + i];
                float f0    = sigf(f_pre[rb + i]        + wx[wb + Hdim + i]);
                float f1    = sigf(f_pre[rb + Hdim + i] + wx[wb + 2 * Hdim + i]);
                float v = csum + (1.0f - f0 - f1) * tanhf(wx[wb + i] + hcadd);
                h[(size_t)p * Hdim + i]   = v;
                hbm[(size_t)p * Hdim + i] = (bf16)v;
            }
        }
    }
}

// ---------------------------------------------------------------------------
// Launcher: bf16 pre-conversions, 1 big GEMM, leaves, then 16 serial levels.
// Workspace (bytes, ~707 MB assumed available):
//   wx f32 | f_pre f32 | fh f32 | xb | hb | fhb | Wwb | Ufb | Uhcb  (bf16)
// ---------------------------------------------------------------------------
extern "C" void kernel_launch(void* const* d_in, const int* in_sizes, int n_in,
                              void* d_out, int out_size, void* d_ws, size_t ws_size,
                              hipStream_t stream) {
    (void)in_sizes; (void)n_in; (void)out_size; (void)ws_size;

    const float* x   = (const float*)d_in[0];   // (N_NODES, 256)
    const float* Ww  = (const float*)d_in[1];   // (768, 256)
    const float* Wb  = (const float*)d_in[2];   // (768,)
    const float* Uhc = (const float*)d_in[3];   // (256, 512)
    const float* Uf  = (const float*)d_in[4];   // (512, 512)
    float* h = (float*)d_out;                   // (N_NODES, 256)

    char* w = (char*)d_ws;
    float* wx    = (float*)w;  w += (size_t)NNODES * WXC * 4;
    float* f_pre = (float*)w;  w += (size_t)32768 * 512 * 4;
    float* fh    = (float*)w;  w += (size_t)32768 * 512 * 4;
    bf16*  xb    = (bf16*)w;   w += (size_t)NNODES * Xdim * 2;
    bf16*  hbm   = (bf16*)w;   w += (size_t)NNODES * Hdim * 2;
    bf16*  fhb   = (bf16*)w;   w += (size_t)32768 * 512 * 2;
    bf16*  Wwb   = (bf16*)w;   w += (size_t)WXC * Xdim * 2;
    bf16*  Ufb   = (bf16*)w;   w += (size_t)512 * 512 * 2;
    bf16*  Uhcb  = (bf16*)w;   w += (size_t)Hdim * 512 * 2;

    dim3 blk(BLK);

    // bf16 pre-conversions (all counts divisible by 8)
    {
        int n8;
        n8 = (NNODES * Xdim) / 8;
        k_cvt<<<dim3((n8 + BLK - 1) / BLK), blk, 0, stream>>>(x, xb, n8);
        n8 = (WXC * Xdim) / 8;
        k_cvt<<<dim3((n8 + BLK - 1) / BLK), blk, 0, stream>>>(Ww, Wwb, n8);
        n8 = (512 * 512) / 8;
        k_cvt<<<dim3((n8 + BLK - 1) / BLK), blk, 0, stream>>>(Uf, Ufb, n8);
        n8 = (Hdim * 512) / 8;
        k_cvt<<<dim3((n8 + BLK - 1) / BLK), blk, 0, stream>>>(Uhc, Uhcb, n8);
    }
    {   // wx = x @ W_w^T + b   (2048 x 6 tiles)
        dim3 g((NNODES + TM - 1) / TM, WXC / TN);
        k_wx<<<g, blk, 0, stream>>>(xb, Wwb, Wb, wx);
    }
    {   // leaves
        k_leaf<<<dim3(1 << TDEPTH), dim3(Hdim), 0, stream>>>(wx, h, hbm);
    }
    for (int d = TDEPTH - 1; d >= 0; --d) {
        int cnt   = 1 << d;
        int start = cnt - 1;
        dim3 g1((cnt + TM - 1) / TM, (2 * Hdim) / TN);   // 512 cols -> 4 tiles
        k_fpre<<<g1, blk, 0, stream>>>(hbm, h, Ufb, f_pre, fh, fhb, start, cnt);
        dim3 g2((cnt + TM - 1) / TM, Hdim / TN);         // 256 cols -> 2 tiles
        k_hout<<<g2, blk, 0, stream>>>(fhb, fh, Uhcb, f_pre, wx, h, hbm, start, cnt);
    }
}
